// BiLSTM_CRF_31439160607362
// MI455X (gfx1250) — compile-verified, matrix operand order checked by reading, not run on hardware
//
#include <hip/hip_runtime.h>
#include <hip/hip_bf16.h>
#include <stdint.h>

#define T_LEN 8192
#define F_DIM 512
#define HD 512
#define G4 (4 * HD)          // 2048 gate rows
#define K_TAGS 10
#define START_TAG 8
#define STOP_TAG 9

typedef __attribute__((ext_vector_type(16))) __bf16 v16bf;
typedef __attribute__((ext_vector_type(8)))  float  v8f;

// ---------------------------------------------------------------- helpers
__device__ __forceinline__ unsigned short f2bf(float f) {
    unsigned int u = __float_as_uint(f);
    u += 0x7FFFu + ((u >> 16) & 1u);          // round-to-nearest-even
    return (unsigned short)(u >> 16);
}
__device__ __forceinline__ float sigmoidf(float x) {
    return 1.0f / (1.0f + __expf(-x));
}

// fp32 -> bf16 conversion
__global__ void k_cvt_bf16(const float* __restrict__ src,
                           unsigned short* __restrict__ dst, long n) {
    long i = (long)blockIdx.x * blockDim.x + threadIdx.x;
    if (i < n) dst[i] = f2bf(src[i]);
}

// combined gate bias  b = b_ih + b_hh  (both directions)
__global__ void k_bias(const float* bif, const float* bhf,
                       const float* bib, const float* bhb,
                       float* bf_, float* bb_) {
    int i = blockIdx.x * blockDim.x + threadIdx.x;
    if (i < G4) { bf_[i] = bif[i] + bhf[i]; bb_[i] = bib[i] + bhb[i]; }
}

// W_hh (2048,512) row-major  ->  W_hh^T (512,2048) so recurrent loop is coalesced
__global__ void k_transpose(const float* __restrict__ w, float* __restrict__ wT) {
    int i = blockIdx.x * blockDim.x + threadIdx.x;
    if (i < G4 * HD) {
        int r = i >> 9;            // row in (2048,512)
        int k = i & 511;
        wT[k * G4 + r] = w[i];
    }
}

// ------------------------------------------------------- WMMA fragment load
// Row-major bf16 source; 16x32 fragment (A: lane=M row, B: lane=N column,
// since W_ih is stored (N,K) row-major).
// Wave32 layout: lanes 0-15 -> rows 0-15 holding K {0..7,16..23},
//                lanes 16-31 -> rows 0-15 holding K {8..15,24..31}.
__device__ __forceinline__ v16bf load_frag16x32(const unsigned short* __restrict__ base,
                                                int row, int kk, int lane) {
    const int koff = (lane >> 4) * 8;
    const unsigned short* p = base + (long)row * F_DIM + kk + koff;
    union { v16bf v; uint4 u[2]; } f;
    f.u[0] = *(const uint4*)(p);        // K chunk 0..7 (or 8..15)
    f.u[1] = *(const uint4*)(p + 16);   // K chunk 16..23 (or 24..31)
    return f.v;
}

// ------------------------------------------------------- xg GEMM (bf16 WMMA)
// xg[dir] = X (8192x512) @ W_ih[dir]^T (512x2048) + bias[dir]
// Register-blocked: each wave computes a 16x64 output strip -> per k-step
// 1 A fragment is reused across 4 B fragments / 4 accumulators (4 WMMAs).
// Tiles: Mt=512, N-blocks=32 per dir, 2 dirs -> 32768 wave-strips.
__global__ void k_xg_gemm(const unsigned short* __restrict__ Xb,
                          const unsigned short* __restrict__ Wfb,
                          const unsigned short* __restrict__ Wbb,
                          const float* __restrict__ bias_f,
                          const float* __restrict__ bias_b,
                          float* __restrict__ xg_f,
                          float* __restrict__ xg_b) {
    const int lane = threadIdx.x & 31;
    const int wave = threadIdx.x >> 5;
    const long tile = (long)blockIdx.x * 8 + wave;
    if (tile >= (long)512 * 64) return;           // wave-uniform guard

    const int mt  = (int)(tile / 64);             // M tile (16 rows)
    const int nb  = (int)(tile % 64);             // 64-wide N block (2 dirs)
    const int dir = nb >> 5;
    const int nbd = nb & 31;

    const unsigned short* W  = dir ? Wbb    : Wfb;
    const float*        bias = dir ? bias_b : bias_f;
    float*              out  = dir ? xg_b   : xg_f;

    const int arow  = mt  * 16 + (lane & 15);     // A row (M)
    const int brow0 = nbd * 64 + (lane & 15);     // first W_ih row (N col)

    v8f c0 = {}, c1 = {}, c2 = {}, c3 = {};
#pragma unroll
    for (int kk = 0; kk < F_DIM; kk += 32) {
        if (kk + 32 < F_DIM) {
            __builtin_prefetch(Xb + (long)arow  * F_DIM + kk + 32, 0, 0);
            __builtin_prefetch(W  + (long)brow0 * F_DIM + kk + 32, 0, 0);
        }
        const v16bf a  = load_frag16x32(Xb, arow,      kk, lane);
        const v16bf b0 = load_frag16x32(W, brow0 +  0, kk, lane);
        const v16bf b1 = load_frag16x32(W, brow0 + 16, kk, lane);
        const v16bf b2 = load_frag16x32(W, brow0 + 32, kk, lane);
        const v16bf b3 = load_frag16x32(W, brow0 + 48, kk, lane);
        c0 = __builtin_amdgcn_wmma_f32_16x16x32_bf16(false, a, false, b0, (short)0, c0, false, false);
        c1 = __builtin_amdgcn_wmma_f32_16x16x32_bf16(false, a, false, b1, (short)0, c1, false, false);
        c2 = __builtin_amdgcn_wmma_f32_16x16x32_bf16(false, a, false, b2, (short)0, c2, false, false);
        c3 = __builtin_amdgcn_wmma_f32_16x16x32_bf16(false, a, false, b3, (short)0, c3, false, false);
    }

    // C/D layout: VGPR r -> M = r + 8*(lane>=16), N = lane & 15
    const int mbase = mt * 16 + ((lane >> 4) ? 8 : 0);
    const int col0  = nbd * 64 + (lane & 15);
    v8f acc[4] = {c0, c1, c2, c3};
#pragma unroll
    for (int s = 0; s < 4; ++s) {
        const int col = col0 + 16 * s;
        const float bn = bias[col];
#pragma unroll
        for (int r = 0; r < 8; ++r)
            out[(long)(mbase + r) * G4 + col] = acc[s][r] + bn;
    }
}

// ------------------------------------------------ recurrent LSTM scan (seq T)
// One 1024-thread workgroup per direction (both run concurrently). h in LDS,
// W_hh^T streamed coalesced from L2 (8 MB total, L2-resident).
// Thread tid owns gate rows {tid, tid+1024}:
//   tid <  512: i_j (j=tid)      and g_j
//   tid >= 512: f_j (j=tid-512)  and o_j  (this half owns c_j)
__global__ void k_lstm(const float* __restrict__ xg_f, const float* __restrict__ xg_b,
                       const float* __restrict__ wT_f, const float* __restrict__ wT_b,
                       float* __restrict__ h_f, float* __restrict__ h_b) {
    const int dir = blockIdx.x;
    const float* xg = dir ? xg_b : xg_f;
    const float* wT = dir ? wT_b : wT_f;
    float*       hs = dir ? h_b  : h_f;

    __shared__ float sh_h[HD];
    __shared__ float sh_ig[HD];
    const int tid = threadIdx.x;

    if (tid < HD) sh_h[tid] = 0.0f;
    float cst = 0.0f;                     // cell state (valid for tid >= 512)
    __syncthreads();

    for (int s = 0; s < T_LEN; ++s) {
        const int t = dir ? (T_LEN - 1 - s) : s;

        float a0 = 0.0f, a1 = 0.0f;
        const float* w0 = wT + tid;           // column tid of W_hh^T
        const float* w1 = wT + tid + 1024;    // column tid+1024
#pragma unroll 4
        for (int k = 0; k < HD; ++k) {
            const float hk = sh_h[k];
            a0 = fmaf(w0[k * G4], hk, a0);
            a1 = fmaf(w1[k * G4], hk, a1);
        }
        const float* x = xg + (long)t * G4;
        const float g0 = x[tid]        + a0;  // i or f pre-activation
        const float g1 = x[tid + 1024] + a1;  // g or o pre-activation
        __syncthreads();                      // all reads of sh_h done

        if (tid < HD)
            sh_ig[tid] = sigmoidf(g0) * tanhf(g1);   // i * g
        __syncthreads();

        if (tid >= HD) {
            const int j = tid - HD;
            const float fgate = sigmoidf(g0);
            const float ogate = sigmoidf(g1);
            cst = fgate * cst + sh_ig[j];
            const float h = ogate * tanhf(cst);
            sh_h[j] = h;
            hs[(long)t * HD + j] = h;
        }
        __syncthreads();
    }
}

// -------------------------------------------------- emission linear (K=10)
__global__ void k_feats(const float* __restrict__ hf, const float* __restrict__ hb,
                        const float* __restrict__ wlin, const float* __restrict__ blin,
                        float* __restrict__ feats) {
    int i = blockIdx.x * blockDim.x + threadIdx.x;
    if (i >= T_LEN * K_TAGS) return;
    const int t = i / K_TAGS, k = i % K_TAGS;
    const float* w   = wlin + k * (2 * HD);
    const float* hft = hf + (long)t * HD;
    const float* hbt = hb + (long)t * HD;
    float acc = blin[k];
    for (int j = 0; j < HD; ++j) acc = fmaf(hft[j], w[j],      acc);
    for (int j = 0; j < HD; ++j) acc = fmaf(hbt[j], w[HD + j], acc);
    feats[i] = acc;
}

// -------------------------------------------- Viterbi forward + backtrack
__global__ void k_viterbi(const float* __restrict__ feats,
                          const float* __restrict__ trans,
                          int* __restrict__ bptrs,
                          float* __restrict__ out) {
    __shared__ float fv[K_TAGS];
    __shared__ float tr[K_TAGS * K_TAGS];
    const int tid = threadIdx.x;                  // 32 threads, 1 wave

    for (int i = tid; i < K_TAGS * K_TAGS; i += 32) tr[i] = trans[i];
    if (tid < K_TAGS) fv[tid] = (tid == START_TAG) ? 0.0f : -10000.0f;
    __syncthreads();

    for (int t = 0; t < T_LEN; ++t) {
        float best = -3.4e38f; int bp = 0;
        if (tid < K_TAGS) {
            for (int p = 0; p < K_TAGS; ++p) {
                const float sc = fv[p] + tr[tid * K_TAGS + p];
                if (sc > best) { best = sc; bp = p; }
            }
            bptrs[t * K_TAGS + tid] = bp;
        }
        __syncthreads();
        if (tid < K_TAGS) fv[tid] = best + feats[t * K_TAGS + tid];
        __syncthreads();
    }

    if (tid == 0) {
        float best = -3.4e38f; int bl = 0;
        for (int k = 0; k < K_TAGS; ++k) {
            const float sc = fv[k] + tr[STOP_TAG * K_TAGS + k];
            if (sc > best) { best = sc; bl = k; }
        }
        out[0] = best;                            // path_score
        int tag = bl;
        for (int t = T_LEN - 1; t >= 0; --t) {    // best_path[t]
            out[1 + t] = (float)tag;
            tag = bptrs[t * K_TAGS + tag];
        }
    }
}

// ---------------------------------------------------------------- launcher
extern "C" void kernel_launch(void* const* d_in, const int* in_sizes, int n_in,
                              void* d_out, int out_size, void* d_ws, size_t ws_size,
                              hipStream_t stream) {
    const float* sentence = (const float*)d_in[0];
    const float* w_ih_f   = (const float*)d_in[1];
    const float* w_hh_f   = (const float*)d_in[2];
    const float* b_ih_f   = (const float*)d_in[3];
    const float* b_hh_f   = (const float*)d_in[4];
    const float* w_ih_b   = (const float*)d_in[5];
    const float* w_hh_b   = (const float*)d_in[6];
    const float* b_ih_b   = (const float*)d_in[7];
    const float* b_hh_b   = (const float*)d_in[8];
    const float* w_lin    = (const float*)d_in[9];
    const float* b_lin    = (const float*)d_in[10];
    const float* trans    = (const float*)d_in[11];
    float* out = (float*)d_out;

    // ---- workspace carve-up (256B aligned)
    char* ws = (char*)d_ws;
    size_t off = 0;
    auto carve = [&](size_t bytes) -> char* {
        char* p = ws + off;
        off = (off + bytes + 255) & ~(size_t)255;
        return p;
    };
    unsigned short* Xb     = (unsigned short*)carve((size_t)T_LEN * F_DIM * 2);
    unsigned short* Wfb    = (unsigned short*)carve((size_t)G4 * F_DIM * 2);
    unsigned short* Wbb    = (unsigned short*)carve((size_t)G4 * F_DIM * 2);
    float* bias_f          = (float*)carve((size_t)G4 * 4);
    float* bias_b          = (float*)carve((size_t)G4 * 4);
    float* wT_f            = (float*)carve((size_t)HD * G4 * 4);
    float* wT_b            = (float*)carve((size_t)HD * G4 * 4);
    float* xg_f            = (float*)carve((size_t)T_LEN * G4 * 4);
    float* xg_b            = (float*)carve((size_t)T_LEN * G4 * 4);
    float* h_f             = (float*)carve((size_t)T_LEN * HD * 4);
    float* h_b             = (float*)carve((size_t)T_LEN * HD * 4);
    float* feats           = (float*)carve((size_t)T_LEN * K_TAGS * 4);
    int*   bptrs           = (int*)  carve((size_t)T_LEN * K_TAGS * 4);
    (void)ws_size; (void)n_in; (void)in_sizes; (void)out_size;

    // ---- stage 0: precision conversion / weight prep
    {
        long n = (long)T_LEN * F_DIM;
        k_cvt_bf16<<<(n + 255) / 256, 256, 0, stream>>>(sentence, Xb, n);
        n = (long)G4 * F_DIM;
        k_cvt_bf16<<<(n + 255) / 256, 256, 0, stream>>>(w_ih_f, Wfb, n);
        k_cvt_bf16<<<(n + 255) / 256, 256, 0, stream>>>(w_ih_b, Wbb, n);
        k_bias<<<(G4 + 255) / 256, 256, 0, stream>>>(b_ih_f, b_hh_f, b_ih_b, b_hh_b,
                                                     bias_f, bias_b);
        k_transpose<<<(G4 * HD + 255) / 256, 256, 0, stream>>>(w_hh_f, wT_f);
        k_transpose<<<(G4 * HD + 255) / 256, 256, 0, stream>>>(w_hh_b, wT_b);
    }

    // ---- stage 1: input-gate GEMM (bf16 WMMA, 16x64 strips), both directions
    // 512 M-tiles * 64 N-blocks = 32768 wave-strips / 8 waves per block
    k_xg_gemm<<<4096, 256, 0, stream>>>(Xb, Wfb, Wbb, bias_f, bias_b, xg_f, xg_b);

    // ---- stage 2: recurrent scans (fwd & bwd run concurrently, 1 WGP each)
    k_lstm<<<2, 1024, 0, stream>>>(xg_f, xg_b, wT_f, wT_b, h_f, h_b);

    // ---- stage 3: emission scores
    k_feats<<<(T_LEN * K_TAGS + 255) / 256, 256, 0, stream>>>(h_f, h_b, w_lin, b_lin, feats);

    // ---- stage 4: Viterbi forward scan + backtrack
    k_viterbi<<<1, 32, 0, stream>>>(feats, trans, bptrs, out);
}
